// TimeSeriesDecoder_8727373546141
// MI455X (gfx1250) — matrix operand order repaired and checked by reading, compiled, not run on hardware
//
#include <hip/hip_runtime.h>
#include <hip/hip_bf16.h>

typedef __attribute__((ext_vector_type(16))) _Float16 v16h;
typedef __attribute__((ext_vector_type(8)))  _Float16 v8h;
typedef __attribute__((ext_vector_type(8)))  float    v8f;

#define Bc   256
#define Lc   128
#define Hc   256
#define Sc   512
#define Oc   64
#define NWG  32

// ---------------------------------------------------------------------------
// helpers
// ---------------------------------------------------------------------------
__device__ __forceinline__ float fsig(float x)  { return 1.0f / (1.0f + __expf(-x)); }
__device__ __forceinline__ float ftanh(float x) { return 1.0f - 2.0f / (__expf(2.0f * x) + 1.0f); }

// A-fragment (16x32, f16): lane L -> row m0+(L&15); halves per documented layout.
// Source LDS tile is row-major [rows][256].
__device__ __forceinline__ v16h loadA(const _Float16* p, int m0, int kb, int lane) {
    int m = m0 + (lane & 15), gl = lane >> 4;
    const _Float16* q = p + m * 256 + kb * 32 + gl * 8;
    v16h r;
    ((v8h*)&r)[0] = *(const v8h*)q;          // k = kb*32 + gl*8 + 0..7
    ((v8h*)&r)[1] = *(const v8h*)(q + 16);   // k = kb*32 + 16 + gl*8 + 0..7
    return r;
}

// B-fragment (32x16, f16): lane L -> col cbase+(L&15); lanes 0-15 hold K 0..7,16..23,
// lanes 16-31 hold K 8..15,24..31.  Weights staged in LDS as [col][256] row-major.
__device__ __forceinline__ v16h loadB(const _Float16* p, int cbase, int kb, int lane) {
    int c = cbase + (lane & 15), gl = lane >> 4;
    const _Float16* q = p + c * 256 + kb * 32 + gl * 8;
    v16h r;
    ((v8h*)&r)[0] = *(const v8h*)q;
    ((v8h*)&r)[1] = *(const v8h*)(q + 16);
    return r;
}

// acc[g] += A(m0, :) * W(g*32+j0 .. +16, :)^T over K=256  (8 k-steps, 4 gates)
__device__ __forceinline__ void mm4(v8f acc[4], const _Float16* a_lds, const _Float16* w_lds,
                                    int m0, int j0, int lane) {
#pragma unroll
    for (int kb = 0; kb < 8; ++kb) {
        v16h a = loadA(a_lds, m0, kb, lane);
#pragma unroll
        for (int g = 0; g < 4; ++g) {
            v16h b = loadB(w_lds, g * 32 + j0, kb, lane);
            acc[g] = __builtin_amdgcn_wmma_f32_16x16x32_f16(false, a, false, b,
                                                            (short)0, acc[g], false, false);
        }
    }
}

// gate nonlinearity + cell update + f16 store of h to global (C-layout aware)
__device__ __forceinline__ void lstm_update(v8f acc[4], float c[8],
                                            _Float16* __restrict__ hg,
                                            int rowb, int colh) {
#pragma unroll
    for (int v = 0; v < 8; ++v) {
        float iv = fsig(acc[0][v]);
        float fv = fsig(acc[1][v]);
        float gv = ftanh(acc[2][v]);
        float ov = fsig(acc[3][v]);
        c[v] = fv * c[v] + iv * gv;
        float hv = ov * ftanh(c[v]);
        hg[(rowb + v) * Hc + colh] = (_Float16)hv;
    }
}

// stage 128 gate-rows of a [1024][256] f32 weight into LDS [128][256] f16
__device__ __forceinline__ void stage_w(const float* __restrict__ g, _Float16* lds, int jb) {
    for (int idx = threadIdx.x; idx < 128 * 256; idx += 256) {
        int c = idx >> 8, k = idx & 255;
        int R = (c >> 5) * 256 + jb * 32 + (c & 31);
        lds[c * 256 + k] = (_Float16)g[R * 256 + k];
    }
}

// synchronous copy 64x256 f16 global -> LDS (used only in the preamble)
__device__ __forceinline__ void stage_rows(const _Float16* __restrict__ g, _Float16* lds, int rb) {
    const uint4* src = (const uint4*)(g + rb * 64 * 256);
    uint4* dst = (uint4*)lds;
    for (int i = threadIdx.x; i < 64 * 256 / 8; i += 256) dst[i] = src[i];
}

// async copy 64x256 f16 global -> LDS: 8 GLOBAL_LOAD_ASYNC_TO_LDS_B128 per wave,
// tracked with ASYNCcnt (direct fabric->LDS, no VGPR round-trip)
__device__ __forceinline__ void stage_rows_async(const _Float16* __restrict__ g,
                                                 _Float16* lds, int rb) {
    const char* src = (const char*)(g + rb * 64 * 256);
    unsigned dst = (unsigned)(uintptr_t)lds;   // low 32 bits of generic ptr = LDS offset
#pragma unroll
    for (int i = threadIdx.x; i < 2048; i += 256) {
        asm volatile("global_load_async_to_lds_b128 %0, %1, off"
                     :: "v"(dst + (unsigned)i * 16u), "v"(src + (size_t)i * 16)
                     : "memory");
    }
}

__device__ __forceinline__ void wait_async_le8() {
    asm volatile("s_wait_asynccnt 0x8" ::: "memory");
}
__device__ __forceinline__ void wait_async_0() {
    asm volatile("s_wait_asynccnt 0x0" ::: "memory");
}

__device__ __forceinline__ void zero_tile(_Float16* lds) {
    uint4 z = {0u, 0u, 0u, 0u};
    uint4* dst = (uint4*)lds;
    for (int i = threadIdx.x; i < 64 * 256 / 8; i += 256) dst[i] = z;
}

// grid-wide barrier: monotonic counter, arrive + spin (32 co-resident WGs).
// Every thread fences its own global stores before arrival.
__device__ __forceinline__ void gsync(unsigned* ctr, unsigned target) {
    __threadfence();
    __syncthreads();
    if (threadIdx.x == 0) {
        __hip_atomic_fetch_add(ctr, 1u, __ATOMIC_RELEASE, __HIP_MEMORY_SCOPE_AGENT);
        while (__hip_atomic_load(ctr, __ATOMIC_ACQUIRE, __HIP_MEMORY_SCOPE_AGENT) < target)
            __builtin_amdgcn_s_sleep(2);
    }
    __syncthreads();
}

// ---------------------------------------------------------------------------
// preamble: FC + ReLU -> f16, zero layer-1 state (parity 0), reset barrier ctr
// ---------------------------------------------------------------------------
__global__ void __launch_bounds__(256) fc_relu_kernel(const float* __restrict__ z,
                                                      const float* __restrict__ fcw,
                                                      const float* __restrict__ fcb,
                                                      _Float16* __restrict__ h0,
                                                      _Float16* __restrict__ hBg,
                                                      unsigned* __restrict__ ctr) {
    int b = blockIdx.x, c = threadIdx.x;
    float s = fcb[c];
#pragma unroll 4
    for (int l = 0; l < Lc; ++l) s = fmaf(z[b * Lc + l], fcw[c * Lc + l], s);
    h0[b * Hc + c]  = (_Float16)fmaxf(s, 0.0f);
    hBg[b * Hc + c] = (_Float16)0.0f;          // hB parity-0 must start at zero
    if (b == 0 && c == 0) *ctr = 0u;
}

// ---------------------------------------------------------------------------
// persistent fused 2-layer LSTM + output projection
// grid = 32 WGs (rb 0..3 x jb 0..7), 256 threads (8 waves)
// ---------------------------------------------------------------------------
__global__ void __launch_bounds__(256)
lstm_persistent(const _Float16* __restrict__ h0g,
                _Float16* __restrict__ hAg,   // 2 x [256][256] f16 (parity buffers)
                _Float16* __restrict__ hBg,   // 2 x [256][256] f16
                unsigned* __restrict__ ctr,
                const float* __restrict__ w_ih0, const float* __restrict__ w_hh0,
                const float* __restrict__ b_ih0, const float* __restrict__ b_hh0,
                const float* __restrict__ w_ih1, const float* __restrict__ w_hh1,
                const float* __restrict__ b_ih1, const float* __restrict__ b_hh1,
                const float* __restrict__ out_w, const float* __restrict__ out_b,
                float* __restrict__ outp) {
    extern __shared__ char smem[];
    _Float16* W0  = (_Float16*)smem;        // w_hh0 slice  [128][256]  64KB
    _Float16* W1i = W0  + 128 * 256;        // w_ih1 slice               64KB
    _Float16* W1h = W1i + 128 * 256;        // w_ih0 (temp) then w_hh1   64KB
    _Float16* hAl = W1h + 128 * 256;        // layer-0 h rows [64][256]  32KB
    _Float16* hBl = hAl + 64 * 256;         // layer-1 h rows            32KB
    _Float16* Wo  = hBl + 64 * 256;         // out_w slice [16][256]      8KB

    const int tid = threadIdx.x;
    const int rb = blockIdx.x >> 3, jb = blockIdx.x & 7;
    const int wv = tid >> 5, lane = tid & 31;
    const int ln = lane & 15, gl = lane >> 4;
    const int m0 = (wv >> 1) * 16, j0 = (wv & 1) * 16;

    // ---- stage weights (f32 -> f16) + FC activations ----
    stage_w(w_hh0, W0, jb);
    stage_w(w_ih1, W1i, jb);
    stage_w(w_ih0, W1h, jb);                 // temporary: needed only for xg0
    stage_rows(h0g, hBl, rb);                // temporary: h0 tile
    if (jb < 4) {                            // out_w rows jb*16..+16
        for (int idx = tid; idx < 16 * 256; idx += 256) {
            int r = idx >> 8, k = idx & 255;
            Wo[r * 256 + k] = (_Float16)out_w[(jb * 16 + r) * 256 + k];
        }
    }
    __syncthreads();

    // ---- precompute xg0 (constant over all timesteps) into registers ----
    v8f xg[4];
    float b1v[4];
#pragma unroll
    for (int g = 0; g < 4; ++g) {
        int col = g * 256 + jb * 32 + j0 + ln;
        float bv = b_ih0[col] + b_hh0[col];
        xg[g]  = (v8f){bv, bv, bv, bv, bv, bv, bv, bv};
        b1v[g] = b_ih1[col] + b_hh1[col];
    }
    mm4(xg, hBl, W1h, m0, j0, lane);         // xg0 += h0 * w_ih0^T
    float obv = 0.0f;
    if (jb < 4 && wv < 4) obv = out_b[jb * 16 + ln];
    __syncthreads();

    // ---- reclaim temp LDS: stage w_hh1, zero h tiles (t = -1 state) ----
    stage_w(w_hh1, W1h, jb);
    zero_tile(hAl);
    zero_tile(hBl);
    __syncthreads();

    float cA[8] = {0, 0, 0, 0, 0, 0, 0, 0};
    float cB[8] = {0, 0, 0, 0, 0, 0, 0, 0};
    const int rowb = rb * 64 + m0 + 8 * gl;  // C/D layout: m = v + 8*(lane/16)
    const int colh = jb * 32 + j0 + ln;
    unsigned epoch = 0;

    // ---- fused supersteps: layer0(t=s) | layer1(t=s-1) | proj(t=s-2) ----
    // Per superstep 16 async B128 copies are in flight: first 8 = hA tile,
    // last 8 = hB tile.  Async loads complete in order, so ASYNCcnt<=8 means
    // the hA tile landed; the hB wait is deferred past the whole layer-0 phase.
    for (int s = 0; s <= Sc; ++s) {
        _Float16* hAw = hAg + (s & 1) * (Bc * Hc);
        _Float16* hBw = hBg + (s & 1) * (Bc * Hc);

        wait_async_le8();                                // my hA copies done
        __syncthreads();                                 // all waves' hA done

        if (s < Sc) {                                    // layer 0, step s
            v8f acc[4] = {xg[0], xg[1], xg[2], xg[3]};
            mm4(acc, hAl, W0, m0, j0, lane);             // + hA_{s-1} * w_hh0^T
            lstm_update(acc, cA, hAw, rowb, colh);
        }

        wait_async_0();                                  // hB copies done
        __syncthreads();

        if (s >= 1) {                                    // layer 1, step s-1
            v8f acc[4];
#pragma unroll
            for (int g = 0; g < 4; ++g)
                acc[g] = (v8f){b1v[g], b1v[g], b1v[g], b1v[g],
                               b1v[g], b1v[g], b1v[g], b1v[g]};
            mm4(acc, hAl, W1i, m0, j0, lane);            // + hA_{s-1} * w_ih1^T
            mm4(acc, hBl, W1h, m0, j0, lane);            // + hB_{s-2} * w_hh1^T
            lstm_update(acc, cB, hBw, rowb, colh);
        }
        if (s >= 2 && jb < 4 && wv < 4) {                // out proj, step s-2
            int t = s - 2;
            v8f acc = {obv, obv, obv, obv, obv, obv, obv, obv};
#pragma unroll
            for (int kb = 0; kb < 8; ++kb) {
                v16h a = loadA(hBl, wv * 16, kb, lane);  // hB_{s-2} rows
                v16h b = loadB(Wo, 0, kb, lane);
                acc = __builtin_amdgcn_wmma_f32_16x16x32_f16(false, a, false, b,
                                                             (short)0, acc, false, false);
            }
            int colo = jb * 16 + ln;
            int rbase = rb * 64 + wv * 16 + 8 * gl;
#pragma unroll
            for (int v = 0; v < 8; ++v)
                outp[((rbase + v) * Sc + t) * Oc + colo] = acc[v];
        }

        ++epoch;
        gsync(ctr, epoch * NWG);                         // publish hA_s, hB_{s-1}
        stage_rows_async(hAw, hAl, rb);                  // hAl <- hA_s   (async)
        stage_rows_async(hBw, hBl, rb);                  // hBl <- hB_{s-1} (async)
    }

    // final projection: t = S-1 (hBl holds hB_{S-1} after the last staging)
    wait_async_0();
    __syncthreads();
    if (jb < 4 && wv < 4) {
        v8f acc = {obv, obv, obv, obv, obv, obv, obv, obv};
#pragma unroll
        for (int kb = 0; kb < 8; ++kb) {
            v16h a = loadA(hBl, wv * 16, kb, lane);
            v16h b = loadB(Wo, 0, kb, lane);
            acc = __builtin_amdgcn_wmma_f32_16x16x32_f16(false, a, false, b,
                                                         (short)0, acc, false, false);
        }
        int colo = jb * 16 + ln;
        int rbase = rb * 64 + wv * 16 + 8 * gl;
#pragma unroll
        for (int v = 0; v < 8; ++v)
            outp[((rbase + v) * Sc + (Sc - 1)) * Oc + colo] = acc[v];
    }
}

// ---------------------------------------------------------------------------
extern "C" void kernel_launch(void* const* d_in, const int* in_sizes, int n_in,
                              void* d_out, int out_size, void* d_ws, size_t ws_size,
                              hipStream_t stream) {
    (void)in_sizes; (void)n_in; (void)out_size; (void)ws_size;
    const float* z     = (const float*)d_in[0];
    const float* fc_w  = (const float*)d_in[1];
    const float* fc_b  = (const float*)d_in[2];
    const float* w_ih0 = (const float*)d_in[3];
    const float* w_hh0 = (const float*)d_in[4];
    const float* b_ih0 = (const float*)d_in[5];
    const float* b_hh0 = (const float*)d_in[6];
    const float* w_ih1 = (const float*)d_in[7];
    const float* w_hh1 = (const float*)d_in[8];
    const float* b_ih1 = (const float*)d_in[9];
    const float* b_hh1 = (const float*)d_in[10];
    const float* out_w = (const float*)d_in[11];
    const float* out_b = (const float*)d_in[12];
    float* out = (float*)d_out;

    char* ws = (char*)d_ws;
    unsigned*  ctr = (unsigned*)ws;                              // barrier counter
    _Float16*  h0  = (_Float16*)(ws + 512);                      // 256*256 f16
    _Float16*  hA  = (_Float16*)(ws + 512 + 131072);             // 2x 256*256 f16
    _Float16*  hB  = (_Float16*)(ws + 512 + 131072 + 262144);    // 2x 256*256 f16

    fc_relu_kernel<<<Bc, Hc, 0, stream>>>(z, fc_w, fc_b, h0, hB, ctr);

    const int SMEM = (3 * 128 * 256 + 2 * 64 * 256 + 16 * 256) * 2;  // 270336 B
    hipFuncSetAttribute((const void*)lstm_persistent,
                        hipFuncAttributeMaxDynamicSharedMemorySize, SMEM);
    lstm_persistent<<<NWG, 256, SMEM, stream>>>(h0, hA, hB, ctr,
                                                w_ih0, w_hh0, b_ih0, b_hh0,
                                                w_ih1, w_hh1, b_ih1, b_hh1,
                                                out_w, out_b, out);
}
